// CharLevelEncoder_20787641713006
// MI455X (gfx1250) — compile-verified, matrix operand order checked by reading, not run on hardware
//
#include <hip/hip_runtime.h>
#include <hip/hip_bf16.h>
#include <math.h>

// Problem constants (match reference)
#define T_CHARS   131072
#define N_WORDS   32768
#define WORD_DIM  1024
#define CHAR_DIM  256
#define HIDDEN    512
#define NUM_EMB   40
#define LIN_IN    (WORD_DIM + HIDDEN)   // 1536
#define KCHUNKS   (WORD_DIM / 32)       // 32 K-chunks of 32

typedef __attribute__((ext_vector_type(16))) __bf16 v16bf;
typedef __attribute__((ext_vector_type(8)))  float  v8f;
typedef __attribute__((ext_vector_type(4)))  float  f4;

// ---------------------------------------------------------------------------
// Kernel 1: h_table[id][j] for the 40 distinct char embeddings.
// gates order (torch): i=[0,H), f=[H,2H) (unused, c0=0), g=[2H,3H), o=[3H,4H)
// ---------------------------------------------------------------------------
__global__ void k_htable(const float* __restrict__ E,
                         const float* __restrict__ W_ih,
                         const float* __restrict__ b_ih,
                         const float* __restrict__ b_hh,
                         float* __restrict__ h_table) {
    int idx = blockIdx.x * blockDim.x + threadIdx.x;     // 40*512
    if (idx >= NUM_EMB * HIDDEN) return;
    int id = idx >> 9;
    int j  = idx & (HIDDEN - 1);
    const float* e  = E + id * CHAR_DIM;
    const float* wi = W_ih + (size_t)j * CHAR_DIM;
    const float* wg = W_ih + (size_t)(2 * HIDDEN + j) * CHAR_DIM;
    const float* wo = W_ih + (size_t)(3 * HIDDEN + j) * CHAR_DIM;
    float si = 0.f, sg = 0.f, so = 0.f;
    for (int k = 0; k < CHAR_DIM; ++k) {
        float x = e[k];
        si = fmaf(x, wi[k], si);
        sg = fmaf(x, wg[k], sg);
        so = fmaf(x, wo[k], so);
    }
    si += b_ih[j]              + b_hh[j];
    sg += b_ih[2 * HIDDEN + j] + b_hh[2 * HIDDEN + j];
    so += b_ih[3 * HIDDEN + j] + b_hh[3 * HIDDEN + j];
    float c = (1.f / (1.f + expf(-si))) * tanhf(sg);
    float h = (1.f / (1.f + expf(-so))) * tanhf(c);
    h_table[idx] = h;
}

// ---------------------------------------------------------------------------
// Kernel 2: hproj[id][m] = b_lin[m] + h_table[id] . W_lin[m][1024:1536]
// ---------------------------------------------------------------------------
__global__ void k_hproj(const float* __restrict__ h_table,
                        const float* __restrict__ W_lin,
                        const float* __restrict__ b_lin,
                        float* __restrict__ hproj) {
    int idx = blockIdx.x * blockDim.x + threadIdx.x;     // 40*1024
    if (idx >= NUM_EMB * WORD_DIM) return;
    int id = idx >> 10;
    int m  = idx & (WORD_DIM - 1);
    const float* h = h_table + id * HIDDEN;
    const float* w = W_lin + (size_t)m * LIN_IN + WORD_DIM;
    float s = b_lin[m];
    for (int k = 0; k < HIDDEN; ++k) s = fmaf(h[k], w[k], s);
    hproj[idx] = s;
}

// ---------------------------------------------------------------------------
// Fragment pre-swizzle: f32 row-major -> bf16 hi/lo in WMMA fragment layout.
//
// A 16x32 (or B 32x16 == 16 W-rows x 32 K) tile, per CDNA5 16-bit layout:
// lane L (l16=L%16, half=L/16) holds elements 0..7 = K[kbase..kbase+7] and
// 8..15 = K[kbase+16..kbase+23], kbase = chunk*32 + 8*half, of row l16.
//
// Storage per tile: 2048 B = [1024 B hi][1024 B lo], lane L at byte L*32.
// Tile index = rowblock * KCHUNKS + chunk. One thread per (tile, lane).
// ---------------------------------------------------------------------------
__global__ __launch_bounds__(256) void k_cvt_frag(const float* __restrict__ src,
                                                  long ld,            // row stride (elems)
                                                  char* __restrict__ dst,
                                                  int nthreads) {
    int idx = blockIdx.x * blockDim.x + threadIdx.x;
    if (idx >= nthreads) return;
    int L    = idx & 31;
    int frag = idx >> 5;
    int c    = frag & (KCHUNKS - 1);
    int rb   = frag / KCHUNKS;
    int row  = rb * 16 + (L & 15);
    int kbase = c * 32 + 8 * (L >> 4);
    const float* p = src + (size_t)row * ld + kbase;
    f4 a0 = *(const f4*)(p);
    f4 a1 = *(const f4*)(p + 4);
    f4 a2 = *(const f4*)(p + 16);
    f4 a3 = *(const f4*)(p + 20);
    float t[16];
#pragma unroll
    for (int i = 0; i < 4; ++i) {
        t[i] = a0[i]; t[4 + i] = a1[i]; t[8 + i] = a2[i]; t[12 + i] = a3[i];
    }
    v16bf hi, lo;
#pragma unroll
    for (int i = 0; i < 16; ++i) {
        __bf16 h = (__bf16)t[i];
        hi[i] = h;
        lo[i] = (__bf16)(t[i] - (float)h);
    }
    char* d = dst + (size_t)frag * 2048 + L * 32;
    *(v16bf*)(d)        = hi;
    *(v16bf*)(d + 1024) = lo;
}

// ---------------------------------------------------------------------------
// Kernel 3: wproj[32768,1024] = word_emb @ W_lin[:, :1024]^T
// Pure-WMMA inner loop on pre-swizzled bf16 hi/lo fragments (3x split,
// f32 accumulate). 4 waves / block, 64x64 per wave = 4x4 tiles.
// ---------------------------------------------------------------------------
__global__ __launch_bounds__(128) void k_wproj(const char* __restrict__ Af,
                                               const char* __restrict__ Bf,
                                               float* __restrict__ C) {
    const int lane = threadIdx.x & 31;
    const int wave = threadIdx.x >> 5;            // 2x2 waves of 64x64
    const int l16  = lane & 15;
    const int half = lane >> 4;
    const int aT = blockIdx.y * 8 + (wave >> 1) * 4;   // base 16-row tile
    const int bT = blockIdx.x * 8 + (wave &  1) * 4;   // base 16-col tile

    v8f acc[4][4] = {};

    const char* aBase = Af + (size_t)aT * KCHUNKS * 2048 + lane * 32;
    const char* bBase = Bf + (size_t)bT * KCHUNKS * 2048 + lane * 32;

    for (int c = 0; c < KCHUNKS; ++c) {
        v16bf ahi[4], alo[4];
#pragma unroll
        for (int m = 0; m < 4; ++m) {
            const char* p = aBase + ((size_t)m * KCHUNKS + c) * 2048;
            ahi[m] = *(const v16bf*)(p);
            alo[m] = *(const v16bf*)(p + 1024);
        }
#pragma unroll
        for (int n = 0; n < 4; ++n) {
            const char* p = bBase + ((size_t)n * KCHUNKS + c) * 2048;
            v16bf bh = *(const v16bf*)(p);
            v16bf bl = *(const v16bf*)(p + 1024);
#pragma unroll
            for (int m = 0; m < 4; ++m) {
                acc[m][n] = __builtin_amdgcn_wmma_f32_16x16x32_bf16(
                    false, ahi[m], false, bh, (short)0, acc[m][n], false, false);
                acc[m][n] = __builtin_amdgcn_wmma_f32_16x16x32_bf16(
                    false, alo[m], false, bh, (short)0, acc[m][n], false, false);
                acc[m][n] = __builtin_amdgcn_wmma_f32_16x16x32_bf16(
                    false, ahi[m], false, bl, (short)0, acc[m][n], false, false);
            }
        }
    }

    // C/D 16x16 f32 layout: lane(l16,half), VGPR v -> M = 8*half + v, N = l16
#pragma unroll
    for (int m = 0; m < 4; ++m) {
#pragma unroll
        for (int n = 0; n < 4; ++n) {
            int col = (bT + n) * 16 + l16;
#pragma unroll
            for (int v = 0; v < 8; ++v) {
                int row = (aT + m) * 16 + half * 8 + v;
                C[(size_t)row * WORD_DIM + col] = acc[m][n][v];
            }
        }
    }
}

// ---------------------------------------------------------------------------
// Kernel 4: out[t] = relu(wproj[word_ids[t]] + hproj[char_ids[t]])
// One block per t; wproj gathers hit L2 (128 MB table < 192 MB L2);
// output streamed with non-temporal stores.
// ---------------------------------------------------------------------------
__global__ __launch_bounds__(256) void k_out(const int* __restrict__ word_ids,
                                             const int* __restrict__ char_ids,
                                             const float* __restrict__ wproj,
                                             const float* __restrict__ hproj,
                                             float* __restrict__ out) {
    int t = blockIdx.x;               // one row of 1024 per block (256 x f4)
    int q = threadIdx.x;
    int wid = word_ids[t];
    int cid = char_ids[t];
    const f4 w = ((const f4*)wproj)[(size_t)wid * 256 + q];
    const f4 h = ((const f4*)hproj)[(size_t)cid * 256 + q];
    f4 r;
#pragma unroll
    for (int i = 0; i < 4; ++i) r[i] = fmaxf(w[i] + h[i], 0.f);
    __builtin_nontemporal_store(r, (f4*)out + (size_t)t * 256 + q);
}

// ---------------------------------------------------------------------------
// Launch
// inputs: 0 word_emb, 1 char_ids, 2 word_ids, 3 E, 4 W_ih, 5 b_ih, 6 b_hh,
//         7 W_lin, 8 b_lin
// d_ws layout: [wproj 128MB][h_table 80KB][hproj 160KB][B_frag 4MB] ~132.3MB
// A_frag (128MB) is staged inside d_out (536MB) and fully overwritten by the
// final kernel -> safe under in-stream ordering, deterministic.
// ---------------------------------------------------------------------------
extern "C" void kernel_launch(void* const* d_in, const int* in_sizes, int n_in,
                              void* d_out, int out_size, void* d_ws, size_t ws_size,
                              hipStream_t stream) {
    const float* word_emb = (const float*)d_in[0];
    const int*   char_ids = (const int*)d_in[1];
    const int*   word_ids = (const int*)d_in[2];
    const float* E        = (const float*)d_in[3];
    const float* W_ih     = (const float*)d_in[4];
    const float* b_ih     = (const float*)d_in[5];
    const float* b_hh     = (const float*)d_in[6];
    const float* W_lin    = (const float*)d_in[7];
    const float* b_lin    = (const float*)d_in[8];
    float* out = (float*)d_out;

    char* ws = (char*)d_ws;
    const size_t WPROJ_B = (size_t)N_WORDS * WORD_DIM * 4;   // 134217728
    float* wproj   = (float*)ws;
    float* h_table = (float*)(ws + WPROJ_B);                 // 40*512*4
    float* hproj   = (float*)(ws + WPROJ_B + 81920);         // 40*1024*4
    char*  B_frag  = ws + WPROJ_B + 81920 + 163840;          // 64*32*2048 = 4MB
    char*  A_frag  = (char*)d_out;                           // 2048*32*2048 = 128MB

    // 0a) swizzle+split word_emb -> A_frag (bf16 hi/lo), into d_out scratch
    {
        int nthr = (N_WORDS / 16) * KCHUNKS * 32;            // 2,097,152
        k_cvt_frag<<<nthr / 256, 256, 0, stream>>>(word_emb, WORD_DIM, A_frag, nthr);
    }
    // 0b) swizzle+split W_lin[:, :1024] -> B_frag
    {
        int nthr = (WORD_DIM / 16) * KCHUNKS * 32;           // 65,536
        k_cvt_frag<<<nthr / 256, 256, 0, stream>>>(W_lin, LIN_IN, B_frag, nthr);
    }

    // 1) 40-entry LSTM table (tiny)
    k_htable<<<(NUM_EMB * HIDDEN + 255) / 256, 256, 0, stream>>>(
        E, W_ih, b_ih, b_hh, h_table);

    // 2) project h table through W_lin[:, 1024:] + b_lin (tiny)
    k_hproj<<<(NUM_EMB * WORD_DIM + 255) / 256, 256, 0, stream>>>(
        h_table, W_lin, b_lin, hproj);

    // 3) big GEMM on pre-swizzled fragments (pure WMMA inner loop)
    dim3 grid(WORD_DIM / 128, N_WORDS / 128);   // (8, 256)
    k_wproj<<<grid, 128, 0, stream>>>(A_frag, B_frag, wproj);

    // 4) gather + add + relu, streaming output (overwrites all of d_out)
    k_out<<<T_CHARS, 256, 0, stream>>>(word_ids, char_ids, wproj, hproj, out);
}
